// GAT_GResNet_11527692222957
// MI455X (gfx1250) — matrix operand order repaired and checked by reading, not compiled, not run on hardware
//
#include <hip/hip_runtime.h>
#include <math.h>

#define NND 50000
#define EED 800000
#define KIN 300
#define KINP 320
#define HH 5
#define DD 32
#define HD 160
#define OUTC 10
#define NEGS 0.2f
#define MAX_KT 10            // max K tiles (KINP/32)

typedef __attribute__((ext_vector_type(16))) __bf16 v16bf;
typedef __attribute__((ext_vector_type(8)))  float  v8f;
typedef __attribute__((ext_vector_type(4)))  unsigned int u32x4;
typedef __attribute__((ext_vector_type(8)))  int    i32x8;
typedef __attribute__((ext_vector_type(4)))  int    i32x4;

#if defined(__has_builtin)
#if __has_builtin(__builtin_amdgcn_tensor_load_to_lds)
#define HAVE_TDM 1
#endif
#endif
#ifndef HAVE_TDM
#define HAVE_TDM 0
#endif

__device__ __forceinline__ unsigned short f2bf(float f) {
    unsigned int u = __float_as_uint(f);
    u += 0x7FFFu + ((u >> 16) & 1u);      // round-to-nearest-even
    return (unsigned short)(u >> 16);
}

__device__ __forceinline__ void atomicMaxF(float* a, float v) {
    if (v >= 0.0f) atomicMax((int*)a, __float_as_int(v));
    else           atomicMin((unsigned int*)a, __float_as_uint(v));
}

#if HAVE_TDM
// TDM: async-copy a 2-D tile of bf16 (dim0_elems per row, `rows` rows, row
// stride `stride_elems`) from global into contiguous LDS at lds_off.
__device__ __forceinline__ void tdm_load_2d(const void* gptr, unsigned lds_off,
                                            unsigned dim0_elems, unsigned rows,
                                            unsigned stride_elems) {
    unsigned long long ga = (unsigned long long)(size_t)gptr;
    u32x4 g0;
    g0[0] = 1u;                                    // count=1 (valid user D#)
    g0[1] = lds_off;                               // lds_addr (bytes)
    g0[2] = (unsigned)ga;                          // global_addr[31:0]
    g0[3] = (unsigned)(ga >> 32) | (2u << 30);     // global_addr[56:32], type=2
    i32x8 g1;
    g1[0] = (int)(1u << 16);                       // wg_mask=0, data_size=1 (2B)
    g1[1] = (int)((dim0_elems & 0xFFFFu) << 16);   // tensor_dim0[15:0]
    g1[2] = (int)(((dim0_elems >> 16) & 0xFFFFu) | ((rows & 0xFFFFu) << 16)); // td0[31:16], td1[15:0]
    g1[3] = (int)(((rows >> 16) & 0xFFFFu) | ((dim0_elems & 0xFFFFu) << 16)); // td1[31:16], tile_dim0
    g1[4] = (int)(rows & 0xFFFFu);                 // tile_dim1 ; tile_dim2=0 (unused)
    g1[5] = (int)stride_elems;                     // tensor_dim0_stride[31:0]
    g1[6] = 0;                                     // stride[47:32]=0, td1_stride=0
    g1[7] = 0;
    i32x4 z4 = {0, 0, 0, 0};
#if __has_include(<hip/amd_detail/amd_gfx1250_TDM.h>)
    i32x8 z8 = {0, 0, 0, 0, 0, 0, 0, 0};
    __builtin_amdgcn_tensor_load_to_lds(g0, g1, z4, z4, z8, 0);
#else
    __builtin_amdgcn_tensor_load_to_lds(g0, g1, z4, z4, 0);
#endif
}
#endif

// ---------------- utility kernels ----------------
__global__ void k_zero(float* __restrict__ p, int n) {
    int i = blockIdx.x * blockDim.x + threadIdx.x;
    if (i < n) p[i] = 0.0f;
}

__global__ void k_deg(const int* __restrict__ dst, float* __restrict__ deg) {
    int e = blockIdx.x * blockDim.x + threadIdx.x;
    if (e < EED) atomicAdd(&deg[dst[e]], 1.0f);
}

__global__ void k_norm(const float* __restrict__ deg, float* __restrict__ nrm) {
    int i = blockIdx.x * blockDim.x + threadIdx.x;
    if (i < NND) { float d = deg[i]; nrm[i] = rsqrtf(d > 1.0f ? d : 1.0f); }
}

__global__ void k_cvt_feat(const float* __restrict__ x, unsigned short* __restrict__ xb) {
    int i = blockIdx.x * blockDim.x + threadIdx.x;      // over N*KINP
    if (i >= NND * KINP) return;
    int n = i / KINP, k = i - n * KINP;
    xb[i] = (k < KIN) ? f2bf(x[n * KIN + k]) : (unsigned short)0;
}

__global__ void k_cvt_h(const float* __restrict__ h, unsigned short* __restrict__ hb, int n) {
    int i = blockIdx.x * blockDim.x + threadIdx.x;
    if (i < n) hb[i] = f2bf(h[i]);
}

// Pack f32 weight [K,Nc] into WMMA B-operand tiles: tile(kt,nt) 32x16;
// lane = (khalf<<4)|ncol holds 16 contiguous bf16: r -> B[kt*32+khalf*16+r][nt*16+ncol]
__global__ void k_pack_b(const float* __restrict__ W, unsigned short* __restrict__ Bp,
                         int K, int Nc, int Kpad, int Npad) {
    int idx = blockIdx.x * blockDim.x + threadIdx.x;
    int ntN = Npad >> 4;
    int total = (Kpad >> 5) * ntN * 512;
    if (idx >= total) return;
    int r    = idx & 15;
    int lane = (idx >> 4) & 31;
    int tile = idx >> 9;
    int kt = tile / ntN, nt = tile - kt * ntN;
    int col = nt * 16 + (lane & 15);
    int k   = kt * 32 + (lane >> 4) * 16 + r;
    float v = (k < K && col < Nc) ? W[k * Nc + col] : 0.0f;
    Bp[idx] = f2bf(v);
}

// -------- WMMA GEMM: C[M,Npad] = A[M,Kpad](bf16) x Bp(packed bf16) + bias --------
// Block = 8 waves = 8 M-tiles; block handles NW (compile-time 1 or 2) 16-wide
// column tiles. B panel staged in LDS once per block via the Tensor Data
// Mover; WMMAs read it back with ds_load_b128; each A fetch feeds NW WMMAs.
template <int NW>
__global__ void k_wmma_gemm(const unsigned short* __restrict__ A,
                            const unsigned short* __restrict__ Bp,
                            const float* __restrict__ bias,
                            float* __restrict__ C,
                            int M, int Kpad, int ntN, int ldc, int ntOffset) {
    __shared__ unsigned short bpan[MAX_KT * 1024];   // up to 2 panels x 10 K-tiles
    const int rowlen = NW * 512;                     // packed elems per K-tile row
    int ktiles = Kpad >> 5;
    int nt0 = blockIdx.y * NW + ntOffset;

#if HAVE_TDM
    if (threadIdx.x == 0) {
        tdm_load_2d(Bp + (size_t)nt0 * 512,
                    (unsigned)(size_t)(void*)bpan,
                    (unsigned)rowlen, (unsigned)ktiles, (unsigned)(ntN * 512));
        __builtin_amdgcn_s_wait_tensorcnt(0);
    }
    __syncthreads();
#else
    for (int idx = threadIdx.x; idx < ktiles * rowlen; idx += blockDim.x) {
        int kt = idx / rowlen, rem = idx - kt * rowlen;
        bpan[idx] = Bp[(size_t)(kt * ntN + nt0) * 512 + rem];
    }
    __syncthreads();
#endif

    int wave = threadIdx.x >> 5;
    int lane = threadIdx.x & 31;
    int mt = blockIdx.x * 8 + wave;
    if (mt >= (M >> 4)) return;

    int hlf = lane >> 4;                             // K-half selector
    const unsigned short* arow = A + (size_t)(mt * 16 + (lane & 15)) * Kpad;

    v8f acc[NW];
#pragma unroll
    for (int j = 0; j < NW; ++j) acc[j] = (v8f){};

    for (int kt = 0; kt < ktiles; ++kt) {
        int kb = kt << 5;
        union { v16bf v; uint4 u[2]; } a;
        // A 16-bit 16x32 layout (matches packed B): lanes 0-15 K[0..16); lanes 16-31 K[16..32)
        a.u[0] = *(const uint4*)(arow + kb + hlf * 16);
        a.u[1] = *(const uint4*)(arow + kb + hlf * 16 + 8);
        if (kt + 1 < ktiles)
            __builtin_prefetch(arow + kb + 32 + hlf * 8, 0, 0);   // global_prefetch_b8
        const unsigned short* bl = bpan + (size_t)kt * rowlen + lane * 16;
#pragma unroll
        for (int j = 0; j < NW; ++j) {
            union { v16bf v; uint4 u[2]; } b;
            b.u[0] = *(const uint4*)(bl + j * 512);
            b.u[1] = *(const uint4*)(bl + j * 512 + 8);
            acc[j] = __builtin_amdgcn_wmma_f32_16x16x32_bf16(
                         false, a.v, false, b.v, (short)0, acc[j], false, false);
        }
    }
    int n = lane & 15;
    int mbase = mt * 16 + hlf * 8;                   // C/D: VGPR v -> M = v + 8*(lane/16)
#pragma unroll
    for (int j = 0; j < NW; ++j) {
        int col = (nt0 + j) * 16 + n;
        float bv = bias ? bias[col] : 0.0f;
#pragma unroll
        for (int v = 0; v < 8; ++v)
            C[(size_t)(mbase + v) * ldc + col] = acc[j][v] + bv;
    }
}

// ---------------- GAT edge-phase kernels (H=5, D=32) ----------------
__global__ void k_elr(const float* __restrict__ feat, const float* __restrict__ al,
                      const float* __restrict__ ar, float* __restrict__ el,
                      float* __restrict__ er) {
    int i = blockIdx.x * blockDim.x + threadIdx.x;    // i = n*H + h
    if (i >= NND * HH) return;
    int h = i % HH;
    const float* f = feat + (size_t)i * DD;
    float sl = 0.f, sr = 0.f;
#pragma unroll
    for (int d = 0; d < DD; ++d) {
        float v = f[d];
        sl += v * al[h * DD + d];
        sr += v * ar[h * DD + d];
    }
    el[i] = sl; er[i] = sr;
}

__global__ void k_init_att(float* __restrict__ emax, float* __restrict__ den, int n) {
    int i = blockIdx.x * blockDim.x + threadIdx.x;
    if (i < n) { emax[i] = __uint_as_float(0xff800000u); den[i] = 0.0f; }
}

__global__ void k_edge_max(const int* __restrict__ src, const int* __restrict__ dst,
                           const float* __restrict__ el, const float* __restrict__ er,
                           float* __restrict__ emax) {
    int i = blockIdx.x * blockDim.x + threadIdx.x;
    if (i >= EED * HH) return;
    int e = i / HH, h = i - e * HH;
    float v = el[src[e] * HH + h] + er[dst[e] * HH + h];
    v = v > 0.f ? v : NEGS * v;
    atomicMaxF(&emax[dst[e] * HH + h], v);
}

__global__ void k_edge_sum(const int* __restrict__ src, const int* __restrict__ dst,
                           const float* __restrict__ el, const float* __restrict__ er,
                           const float* __restrict__ emax, float* __restrict__ den) {
    int i = blockIdx.x * blockDim.x + threadIdx.x;
    if (i >= EED * HH) return;
    int e = i / HH, h = i - e * HH;
    int d = dst[e];
    float v = el[src[e] * HH + h] + er[d * HH + h];
    v = v > 0.f ? v : NEGS * v;
    atomicAdd(&den[d * HH + h], __expf(v - emax[d * HH + h]));
}

__global__ void k_edge_msg(const int* __restrict__ src, const int* __restrict__ dst,
                           const float* __restrict__ el, const float* __restrict__ er,
                           const float* __restrict__ emax, const float* __restrict__ den,
                           const float* __restrict__ feat, float* __restrict__ agg) {
    int i = blockIdx.x * blockDim.x + threadIdx.x;
    if (i >= EED * HH) return;
    int e = i / HH, h = i - e * HH;
    int s = src[e], d = dst[e];
    float v = el[s * HH + h] + er[d * HH + h];
    v = v > 0.f ? v : NEGS * v;
    float alpha = __expf(v - emax[d * HH + h]) / den[d * HH + h];
    const float* fs = feat + (size_t)s * HD + h * DD;
    float* ag = agg + (size_t)d * HD + h * DD;
#pragma unroll 8
    for (int j = 0; j < DD; ++j) atomicAdd(&ag[j], fs[j] * alpha);
}

// fused degree-normalized residual: agg[dst] += norm[dst]*norm[src]*raw[src]
__global__ void k_edge_res(const int* __restrict__ src, const int* __restrict__ dst,
                           const float* __restrict__ nrm, const float* __restrict__ raw,
                           float* __restrict__ agg) {
    int i = blockIdx.x * blockDim.x + threadIdx.x;    // i = e*5 + chunk
    if (i >= EED * HH) return;
    int e = i / HH, c0 = (i - e * HH) * DD;
    int s = src[e], d = dst[e];
    float sc = nrm[s] * nrm[d];
    const float* rs = raw + (size_t)s * HD + c0;
    float* ag = agg + (size_t)d * HD + c0;
#pragma unroll 8
    for (int j = 0; j < DD; ++j) atomicAdd(&ag[j], rs[j] * sc);
}

__global__ void k_relu(float* __restrict__ p, int n) {
    int i = blockIdx.x * blockDim.x + threadIdx.x;
    if (i < n) p[i] = fmaxf(p[i], 0.0f);
}

// ---------------- output layer (H=1, OUT=10, featO padded to ld 16) ----------------
__global__ void k_elr_out(const float* __restrict__ featO, const float* __restrict__ al,
                          const float* __restrict__ ar, float* __restrict__ el,
                          float* __restrict__ er) {
    int n = blockIdx.x * blockDim.x + threadIdx.x;
    if (n >= NND) return;
    const float* f = featO + (size_t)n * 16;
    float sl = 0.f, sr = 0.f;
#pragma unroll
    for (int c = 0; c < OUTC; ++c) { sl += f[c] * al[c]; sr += f[c] * ar[c]; }
    el[n] = sl; er[n] = sr;
}

__global__ void k_edge_max1(const int* __restrict__ src, const int* __restrict__ dst,
                            const float* __restrict__ el, const float* __restrict__ er,
                            float* __restrict__ emax) {
    int e = blockIdx.x * blockDim.x + threadIdx.x;
    if (e >= EED) return;
    float v = el[src[e]] + er[dst[e]];
    v = v > 0.f ? v : NEGS * v;
    atomicMaxF(&emax[dst[e]], v);
}

__global__ void k_edge_sum1(const int* __restrict__ src, const int* __restrict__ dst,
                            const float* __restrict__ el, const float* __restrict__ er,
                            const float* __restrict__ emax, float* __restrict__ den) {
    int e = blockIdx.x * blockDim.x + threadIdx.x;
    if (e >= EED) return;
    int d = dst[e];
    float v = el[src[e]] + er[d];
    v = v > 0.f ? v : NEGS * v;
    atomicAdd(&den[d], __expf(v - emax[d]));
}

__global__ void k_edge_msg1(const int* __restrict__ src, const int* __restrict__ dst,
                            const float* __restrict__ el, const float* __restrict__ er,
                            const float* __restrict__ emax, const float* __restrict__ den,
                            const float* __restrict__ featO, float* __restrict__ out) {
    int e = blockIdx.x * blockDim.x + threadIdx.x;
    if (e >= EED) return;
    int s = src[e], d = dst[e];
    float v = el[s] + er[d];
    v = v > 0.f ? v : NEGS * v;
    float alpha = __expf(v - emax[d]) / den[d];
    const float* fs = featO + (size_t)s * 16;
    float* o = out + (size_t)d * OUTC;
#pragma unroll
    for (int c = 0; c < OUTC; ++c) atomicAdd(&o[c], fs[c] * alpha);
}

// ---------------- host side ----------------
static inline int cdiv(int a, int b) { return (a + b - 1) / b; }

static void launch_gemm(const unsigned short* A, const unsigned short* Bp,
                        const float* bias, float* C, int M, int Kpad, int Npad,
                        int ldc, hipStream_t stream) {
    int ntN = Npad >> 4;
    int pairs = ntN >> 1;
    if (pairs > 0) {
        dim3 grid(cdiv(M >> 4, 8), pairs);
        k_wmma_gemm<2><<<grid, 256, 0, stream>>>(A, Bp, bias, C, M, Kpad, ntN, ldc, 0);
    }
    if (ntN & 1) {
        dim3 grid(cdiv(M >> 4, 8), 1);
        k_wmma_gemm<1><<<grid, 256, 0, stream>>>(A, Bp, bias, C, M, Kpad, ntN, ldc, ntN - 1);
    }
}

static void launch_pack(const float* W, unsigned short* Bp, int K, int Nc,
                        int Kpad, int Npad, hipStream_t stream) {
    int total = (Kpad >> 5) * (Npad >> 4) * 512;
    k_pack_b<<<cdiv(total, 256), 256, 0, stream>>>(W, Bp, K, Nc, Kpad, Npad);
}

static void gat_edge_phase(const int* src, const int* dst, const float* feat,
                           const float* al, const float* ar, float* el, float* er,
                           float* emax, float* den, float* agg, hipStream_t stream) {
    const int B = 256;
    k_elr<<<cdiv(NND * HH, B), B, 0, stream>>>(feat, al, ar, el, er);
    k_init_att<<<cdiv(NND * HH, B), B, 0, stream>>>(emax, den, NND * HH);
    k_edge_max<<<cdiv(EED * HH, B), B, 0, stream>>>(src, dst, el, er, emax);
    k_edge_sum<<<cdiv(EED * HH, B), B, 0, stream>>>(src, dst, el, er, emax, den);
    k_zero<<<cdiv(NND * HD, B), B, 0, stream>>>(agg, NND * HD);
    k_edge_msg<<<cdiv(EED * HH, B), B, 0, stream>>>(src, dst, el, er, emax, den, feat, agg);
}

extern "C" void kernel_launch(void* const* d_in, const int* in_sizes, int n_in,
                              void* d_out, int out_size, void* d_ws, size_t ws_size,
                              hipStream_t stream) {
    const float* features = (const float*)d_in[0];
    const int*   src      = (const int*)d_in[1];
    const int*   dst      = (const int*)d_in[2];
    const float* W0       = (const float*)d_in[3];
    const float* al0      = (const float*)d_in[4];
    const float* ar0      = (const float*)d_in[5];
    const float* W1       = (const float*)d_in[6];
    const float* al1      = (const float*)d_in[7];
    const float* ar1      = (const float*)d_in[8];
    const float* W2       = (const float*)d_in[9];
    const float* al2      = (const float*)d_in[10];
    const float* ar2      = (const float*)d_in[11];
    const float* Wout     = (const float*)d_in[12];
    const float* alout    = (const float*)d_in[13];
    const float* arout    = (const float*)d_in[14];
    const float* Wraw     = (const float*)d_in[15];
    const float* braw     = (const float*)d_in[16];
    float* out = (float*)d_out;

    char* wsp = (char*)d_ws;
    auto alloc = [&](size_t bytes) {
        char* p = wsp;
        wsp += (bytes + 255) & ~(size_t)255;
        return p;
    };
    unsigned short* xbf  = (unsigned short*)alloc((size_t)NND * KINP * 2); // reused as bf16 h
    float* raw   = (float*)alloc((size_t)NND * HD * 4);
    float* feat  = (float*)alloc((size_t)NND * HD * 4);
    float* agg   = (float*)alloc((size_t)NND * HD * 4);
    float* el    = (float*)alloc((size_t)NND * HH * 4);
    float* er    = (float*)alloc((size_t)NND * HH * 4);
    float* emax  = (float*)alloc((size_t)NND * HH * 4);
    float* den   = (float*)alloc((size_t)NND * HH * 4);
    float* deg   = (float*)alloc((size_t)NND * 4);
    float* nrm   = (float*)alloc((size_t)NND * 4);
    float* featO = (float*)alloc((size_t)NND * 16 * 4);
    float* elO   = (float*)alloc((size_t)NND * 4);
    float* erO   = (float*)alloc((size_t)NND * 4);
    float* emaxO = (float*)alloc((size_t)NND * 4);
    float* denO  = (float*)alloc((size_t)NND * 4);
    unsigned short* W0p   = (unsigned short*)alloc((size_t)KINP * HD * 2);
    unsigned short* Wrawp = (unsigned short*)alloc((size_t)KINP * HD * 2);
    unsigned short* W1p   = (unsigned short*)alloc((size_t)HD * HD * 2);
    unsigned short* W2p   = (unsigned short*)alloc((size_t)HD * HD * 2);
    unsigned short* Woutp = (unsigned short*)alloc((size_t)HD * 16 * 2);

    const int B = 256;

    // degrees + norm
    k_zero<<<cdiv(NND, B), B, 0, stream>>>(deg, NND);
    k_deg<<<cdiv(EED, B), B, 0, stream>>>(dst, deg);
    k_norm<<<cdiv(NND, B), B, 0, stream>>>(deg, nrm);

    // bf16 inputs + packed weights
    k_cvt_feat<<<cdiv(NND * KINP, B), B, 0, stream>>>(features, xbf);
    launch_pack(W0,   W0p,   KIN, HD,   KINP, HD, stream);
    launch_pack(Wraw, Wrawp, KIN, HD,   KINP, HD, stream);
    launch_pack(W1,   W1p,   HD,  HD,   HD,   HD, stream);
    launch_pack(W2,   W2p,   HD,  HD,   HD,   HD, stream);
    launch_pack(Wout, Woutp, HD,  OUTC, HD,   16, stream);

    // raw = features @ Wraw + braw ; feat = features @ W0
    launch_gemm(xbf, Wrawp, braw,    raw,  NND, KINP, HD, HD, stream);
    launch_gemm(xbf, W0p,   nullptr, feat, NND, KINP, HD, HD, stream);

    // layer 0: GAT + relu  (h lives in agg)
    gat_edge_phase(src, dst, feat, al0, ar0, el, er, emax, den, agg, stream);
    k_relu<<<cdiv(NND * HD, B), B, 0, stream>>>(agg, NND * HD);

    // layers 1,2: GAT + fused residual + relu
    const float* als[2] = {al1, al2};
    const float* ars[2] = {ar1, ar2};
    unsigned short* Wps[2] = {W1p, W2p};
    for (int l = 0; l < 2; ++l) {
        k_cvt_h<<<cdiv(NND * HD, B), B, 0, stream>>>(agg, xbf, NND * HD);
        launch_gemm(xbf, Wps[l], nullptr, feat, NND, HD, HD, HD, stream);
        gat_edge_phase(src, dst, feat, als[l], ars[l], el, er, emax, den, agg, stream);
        k_edge_res<<<cdiv(EED * HH, B), B, 0, stream>>>(src, dst, nrm, raw, agg);
        k_relu<<<cdiv(NND * HD, B), B, 0, stream>>>(agg, NND * HD);
    }

    // output layer: H=1, OUT=10 (ld 16); mean over 1 head == identity
    k_cvt_h<<<cdiv(NND * HD, B), B, 0, stream>>>(agg, xbf, NND * HD);
    launch_gemm(xbf, Woutp, nullptr, featO, NND, HD, 16, 16, stream);
    k_elr_out<<<cdiv(NND, B), B, 0, stream>>>(featO, alout, arout, elO, erO);
    k_init_att<<<cdiv(NND, B), B, 0, stream>>>(emaxO, denO, NND);
    k_edge_max1<<<cdiv(EED, B), B, 0, stream>>>(src, dst, elO, erO, emaxO);
    k_edge_sum1<<<cdiv(EED, B), B, 0, stream>>>(src, dst, elO, erO, emaxO, denO);
    k_zero<<<cdiv(NND * OUTC, B), B, 0, stream>>>(out, NND * OUTC);
    k_edge_msg1<<<cdiv(EED, B), B, 0, stream>>>(src, dst, elO, erO, emaxO, denO, featO, out);
}
